// PatAttention_21723944583507
// MI455X (gfx1250) — matrix operand order repaired and checked by reading, compiled, not run on hardware
//
#include <hip/hip_runtime.h>
#include <hip/hip_bf16.h>

// Problem constants
constexpr int B_C  = 4;
constexpr int H_C  = 16;
constexpr int N_C  = 128;
constexpr int DK_C = 64;
constexpr int DE_C = 64;

constexpr int LDK = DK_C + 8;   // 72 -> row pitch 144 B (16B aligned)
constexpr int LDN = N_C + 8;    // 136 -> row pitch 272 B (16B aligned)

typedef __attribute__((ext_vector_type(16))) _Float16 v16h;
typedef __attribute__((ext_vector_type(8)))  _Float16 h8;
typedef __attribute__((ext_vector_type(8)))  float    v8f;

__device__ __forceinline__ float leaky(float x) {
  return fmaxf(x, 0.f) + 0.1f * fminf(x, 0.f);
}

// ---------------------------------------------------------------------------
// WMMA helpers (CDNA5 wave32, 16x16x32 f16 -> f32, ISA 7.12.2 layouts).
// Every fragment = 2x ds_load_b128.
// ---------------------------------------------------------------------------
__device__ __forceinline__ v8f wmma16(v16h a, v16h b, v8f c) {
  return __builtin_amdgcn_wmma_f32_16x16x32_f16(false, a, false, b, (short)0, c,
                                                false, false);
}

// A fragment: source row-major [m][k], ld % 8 == 0, k0 % 16 == 0.
__device__ __forceinline__ v16h frag_a_rm(const _Float16* s, int ld, int m0, int k0) {
  const int lane = threadIdx.x & 31;
  const int m = m0 + (lane & 15), half = lane >> 4;
  const _Float16* p = s + m * ld + k0 + half * 8;
  union { v16h v; h8 h[2]; } u;
  u.h[0] = *(const h8*)(p);
  u.h[1] = *(const h8*)(p + 16);
  return u.v;
}

// B fragment: source stored TRANSPOSED [n][k], ld % 8 == 0, k0 % 16 == 0.
__device__ __forceinline__ v16h frag_b_ct(const _Float16* s, int ld, int k0, int n0) {
  const int lane = threadIdx.x & 31;
  const int n = n0 + (lane & 15), half = lane >> 4;
  const _Float16* p = s + n * ld + k0 + half * 8;
  union { v16h v; h8 h[2]; } u;
  u.h[0] = *(const h8*)(p);
  u.h[1] = *(const h8*)(p + 16);
  return u.v;
}

// ---------------------------------------------------------------------------
// K1: edge MLP layer 1 + LeakyReLU.  Per (b,i):
//   hdnK[j][k] (row-major) and hdnVt[k][j] (transposed, for K5's B operand).
// ---------------------------------------------------------------------------
__global__ __launch_bounds__(128) void k1_edge_mlp(
    const float* __restrict__ edges,
    const float* __restrict__ Wk1, const float* __restrict__ bk1,
    const float* __restrict__ Wv1, const float* __restrict__ bv1,
    _Float16* __restrict__ hdnK, _Float16* __restrict__ hdnVt) {
  const int i = blockIdx.x, b = blockIdx.y;
  const int t = threadIdx.x, lane = t & 31, wave = t >> 5;
  __shared__ __align__(16) _Float16 EH[N_C][DE_C + 8];   // [j][c]  (A layout)
  __shared__ __align__(16) _Float16 WK[DK_C][DE_C + 8];  // [k][c]  (B^T layout)
  __shared__ __align__(16) _Float16 WV[DK_C][DE_C + 8];
  __shared__ float BK[DK_C], BV[DK_C];

  const float* eb = edges + (size_t)b * DE_C * N_C * N_C + (size_t)i * N_C;
  for (int c = 0; c < DE_C; ++c) {
    if (c + 1 < DE_C) __builtin_prefetch(eb + (size_t)(c + 1) * N_C * N_C + t, 0, 0);
    EH[t][c] = (_Float16)eb[(size_t)c * N_C * N_C + t];  // coalesced read over j=t
  }
  {
    const float4* wk4 = (const float4*)Wk1;
    const float4* wv4 = (const float4*)Wv1;
    for (int idx = t; idx < DK_C * DE_C / 4; idx += 128) {
      int k = idx >> 4, c0 = (idx & 15) * 4;   // natural [k][c]
      float4 a = wk4[idx], v = wv4[idx];
      WK[k][c0 + 0] = (_Float16)a.x; WK[k][c0 + 1] = (_Float16)a.y;
      WK[k][c0 + 2] = (_Float16)a.z; WK[k][c0 + 3] = (_Float16)a.w;
      WV[k][c0 + 0] = (_Float16)v.x; WV[k][c0 + 1] = (_Float16)v.y;
      WV[k][c0 + 2] = (_Float16)v.z; WV[k][c0 + 3] = (_Float16)v.w;
    }
  }
  if (t < DK_C) { BK[t] = bk1[t]; BV[t] = bv1[t]; }
  __syncthreads();

  const size_t outbase = (size_t)(b * N_C + i) * N_C * DK_C;
  for (int jt = wave * 2; jt < wave * 2 + 2; ++jt) {
    for (int kt = 0; kt < 4; ++kt) {
      v8f aK = {}; v8f aV = {};
#pragma unroll
      for (int kc = 0; kc < 2; ++kc) {
        v16h af = frag_a_rm(&EH[0][0], DE_C + 8, jt * 16, kc * 32);
        aK = wmma16(af, frag_b_ct(&WK[0][0], DE_C + 8, kc * 32, kt * 16), aK);
        aV = wmma16(af, frag_b_ct(&WV[0][0], DE_C + 8, kc * 32, kt * 16), aV);
      }
      int n  = kt * 16 + (lane & 15);             // output channel k
      int mb = jt * 16 + ((lane >> 4) << 3);      // row j base
      h8 pv;                                      // 8 consecutive j for hdnVt
#pragma unroll
      for (int r = 0; r < 8; ++r) {
        float vk = leaky(aK[r] + BK[n]);
        float vv = leaky(aV[r] + BV[n]);
        hdnK[outbase + (size_t)(mb + r) * DK_C + n] = (_Float16)vk;  // [j][k]
        pv[r] = (_Float16)vv;
      }
      *(h8*)(hdnVt + outbase + (size_t)n * N_C + mb) = pv;           // [k][j], 16B
    }
  }
}

// ---------------------------------------------------------------------------
// K2: per (b,h):  g = (1/8)*((q+k) @ Wk2_h) + w_row,  cbase scalars
// ---------------------------------------------------------------------------
__global__ __launch_bounds__(128) void k2_g(
    const float* __restrict__ q, const float* __restrict__ kkey,
    const float* __restrict__ Wk2, const float* __restrict__ bk2,
    const float* __restrict__ rel_u, const float* __restrict__ rel_v,
    _Float16* __restrict__ g, float* __restrict__ cb) {
  const int h = blockIdx.x, b = blockIdx.y;
  const int t = threadIdx.x, lane = t & 31, wave = t >> 5;
  __shared__ __align__(16) _Float16 QK[N_C][LDK];     // [i][d]   (A)
  __shared__ __align__(16) _Float16 W2t[DK_C][LDK];   // [kk][d]  (B^T)
  __shared__ float wrow[DK_C];
  const size_t qb = (size_t)(b * H_C + h) * N_C * DK_C;

  {
    const float4* q4 = (const float4*)(q + qb);
    const float4* k4 = (const float4*)(kkey + qb);
    for (int idx = t; idx < N_C * DK_C / 4; idx += 128) {
      int i = idx >> 4, d0 = (idx & 15) * 4;
      float4 a = q4[idx], bq = k4[idx];
      QK[i][d0 + 0] = (_Float16)(a.x + bq.x);
      QK[i][d0 + 1] = (_Float16)(a.y + bq.y);
      QK[i][d0 + 2] = (_Float16)(a.z + bq.z);
      QK[i][d0 + 3] = (_Float16)(a.w + bq.w);
    }
    const float4* w4 = (const float4*)(Wk2 + (size_t)h * DK_C * DK_C);
    for (int idx = t; idx < DK_C * DK_C / 4; idx += 128) {
      int d = idx >> 4, kk0 = (idx & 15) * 4;
      float4 w = w4[idx];
      W2t[kk0 + 0][d] = (_Float16)w.x;
      W2t[kk0 + 1][d] = (_Float16)w.y;
      W2t[kk0 + 2][d] = (_Float16)w.z;
      W2t[kk0 + 3][d] = (_Float16)w.w;
    }
  }
  if (t < DK_C) {
    float s = 0.f;
    for (int d = 0; d < DK_C; ++d)
      s += rel_v[h * DK_C + d] * Wk2[(size_t)(h * DK_C + d) * DK_C + t];
    wrow[t] = s;
  }
  __syncthreads();

  for (int it = wave * 2; it < wave * 2 + 2; ++it) {
    for (int kt = 0; kt < 4; ++kt) {
      v8f acc = {};
#pragma unroll
      for (int kc = 0; kc < 2; ++kc)
        acc = wmma16(frag_a_rm(&QK[0][0], LDK, it * 16, kc * 32),
                     frag_b_ct(&W2t[0][0], LDK, kc * 32, kt * 16), acc);
      int n  = kt * 16 + (lane & 15);
      int mb = it * 16 + ((lane >> 4) << 3);
#pragma unroll
      for (int r = 0; r < 8; ++r)
        g[qb + (size_t)(mb + r) * DK_C + n] = (_Float16)(0.125f * acc[r] + wrow[n]);
    }
  }
  // cbase[b,h,i] = (1/8)*(q+k)_i . bk2_h + rel_v_h . bk2_h + key_i . rel_u_h
  {
    const int i = t;
    float s = 0.f, s3 = 0.f, c4 = 0.f;
    for (int d = 0; d < DK_C; ++d) {
      float kv = kkey[qb + (size_t)i * DK_C + d];
      float qk = q[qb + (size_t)i * DK_C + d] + kv;
      float bb = bk2[h * DK_C + d];
      s  += qk * bb;
      s3 += kv * rel_u[h * DK_C + d];
      c4 += rel_v[h * DK_C + d] * bb;
    }
    cb[(b * H_C + h) * N_C + i] = 0.125f * s + c4 + s3;
  }
}

// ---------------------------------------------------------------------------
// K3: per (b,i):  s24[h][j] = G(16x64) @ hdnK_i^T(64x128) + cbase
// ---------------------------------------------------------------------------
__global__ __launch_bounds__(128) void k3_s24(
    const _Float16* __restrict__ hdnK, const _Float16* __restrict__ g,
    const float* __restrict__ cb, float* __restrict__ s24) {
  const int i = blockIdx.x, b = blockIdx.y;
  const int t = threadIdx.x, lane = t & 31, wave = t >> 5;
  __shared__ __align__(16) _Float16 HK[N_C][LDK];   // [j][kk]  (B^T: n=j, k=kk)
  __shared__ __align__(16) _Float16 G[H_C][LDK];    // [h][kk]  (A)
  const size_t hb = (size_t)(b * N_C + i) * N_C * DK_C;

  // 16B packed copy: each row of HK = 8 uint4
  for (int idx = t; idx < N_C * DK_C / 8; idx += 128) {
    int j = idx >> 3, kp = idx & 7;
    ((uint4*)&HK[j][0])[kp] = ((const uint4*)(hdnK + hb))[idx];
  }
  for (int idx = t; idx < H_C * DK_C / 8; idx += 128) {
    int h = idx >> 3, kp = idx & 7;
    ((uint4*)&G[h][0])[kp] =
        ((const uint4*)(g + ((size_t)(b * H_C + h) * N_C + i) * DK_C))[kp];
  }
  __syncthreads();

  for (int jt = wave * 2; jt < wave * 2 + 2; ++jt) {
    v8f acc = {};
#pragma unroll
    for (int kc = 0; kc < 2; ++kc)
      acc = wmma16(frag_a_rm(&G[0][0], LDK, 0, kc * 32),
                   frag_b_ct(&HK[0][0], LDK, kc * 32, jt * 16), acc);
    int n  = jt * 16 + (lane & 15);       // j
    int mb = (lane >> 4) << 3;            // h base
#pragma unroll
    for (int r = 0; r < 8; ++r) {
      int h = mb + r;
      s24[((size_t)(b * H_C + h) * N_C + i) * N_C + n] =
          acc[r] + cb[(b * H_C + h) * N_C + i];
    }
  }
}

// ---------------------------------------------------------------------------
// K4: per (b,h): scores = (1/8)*Q@K^T + s24 ; masked softmax ; out1 = P@V
// ---------------------------------------------------------------------------
__global__ __launch_bounds__(256) void k4_attn(
    const float* __restrict__ q, const float* __restrict__ kkey,
    const float* __restrict__ v, const float* __restrict__ mask,
    float* __restrict__ s24, float* __restrict__ o1,
    float* __restrict__ pattn) {
  const int h = blockIdx.x, b = blockIdx.y;
  const int t = threadIdx.x, lane = t & 31, wave = t >> 5;
  // Q|K area (2*128*72 = 18432 f16) is reused for P (128*136 = 17408 f16).
  __shared__ __align__(16) _Float16 smemA[2 * N_C * LDK];
  __shared__ __align__(16) _Float16 Vt[DK_C][LDN];   // [d][j]  (B^T for out1)
  _Float16* Qh = smemA;                 // ld LDK
  _Float16* Kh = smemA + N_C * LDK;     // ld LDK  (B^T for scores: n=j, k=d)
  _Float16* P  = smemA;                 // ld LDN

  const size_t qb = (size_t)(b * H_C + h) * N_C * DK_C;
  {
    const float4* q4 = (const float4*)(q + qb);
    const float4* k4 = (const float4*)(kkey + qb);
    const float4* v4 = (const float4*)(v + qb);
    for (int idx = t; idx < N_C * DK_C / 4; idx += 256) {
      int i = idx >> 4, d0 = (idx & 15) * 4;
      float4 a = q4[idx], bq = k4[idx], vv = v4[idx];
      Qh[i * LDK + d0 + 0] = (_Float16)a.x;  Qh[i * LDK + d0 + 1] = (_Float16)a.y;
      Qh[i * LDK + d0 + 2] = (_Float16)a.z;  Qh[i * LDK + d0 + 3] = (_Float16)a.w;
      Kh[i * LDK + d0 + 0] = (_Float16)bq.x; Kh[i * LDK + d0 + 1] = (_Float16)bq.y;
      Kh[i * LDK + d0 + 2] = (_Float16)bq.z; Kh[i * LDK + d0 + 3] = (_Float16)bq.w;
      Vt[d0 + 0][i] = (_Float16)vv.x; Vt[d0 + 1][i] = (_Float16)vv.y;
      Vt[d0 + 2][i] = (_Float16)vv.z; Vt[d0 + 3][i] = (_Float16)vv.w;
    }
  }
  __syncthreads();

  const size_t sb = (size_t)(b * H_C + h) * N_C * N_C;
  // scores1 tiles (8x8), fused add of s24 -> full scores in s24 buffer
  for (int tt = wave; tt < 64; tt += 8) {
    int it = tt >> 3, jt = tt & 7;
    v8f acc = {};
#pragma unroll
    for (int kc = 0; kc < 2; ++kc)
      acc = wmma16(frag_a_rm(Qh, LDK, it * 16, kc * 32),
                   frag_b_ct(Kh, LDK, kc * 32, jt * 16), acc);
    int n  = jt * 16 + (lane & 15);
    int mb = it * 16 + ((lane >> 4) << 3);
#pragma unroll
    for (int r = 0; r < 8; ++r) {
      size_t idx = sb + (size_t)(mb + r) * N_C + n;
      s24[idx] = 0.125f * acc[r] + s24[idx];
    }
  }
  __threadfence();
  __syncthreads();

  // masked softmax, one row per thread (threads 0..127)
  if (t < N_C) {
    const int i = t;
    const float* srow = s24 + sb + (size_t)i * N_C;
    float mx = -3.0e38f;
    for (int j = 0; j < N_C; ++j) {
      float sv = (mask[b * N_C + j] == 0.f) ? -1.0e12f : srow[j];
      mx = fmaxf(mx, sv);
    }
    float sum = 0.f;
    for (int j = 0; j < N_C; ++j) {
      float sv = (mask[b * N_C + j] == 0.f) ? -1.0e12f : srow[j];
      sum += __expf(sv - mx);
    }
    float inv = 1.f / sum;
    for (int j = 0; j < N_C; ++j) {
      float sv = (mask[b * N_C + j] == 0.f) ? -1.0e12f : srow[j];
      float p  = __expf(sv - mx) * inv;
      pattn[sb + (size_t)i * N_C + j] = p;
      P[i * LDN + j] = (_Float16)p;
    }
  }
  __syncthreads();

  // out1 = P(128x128) @ V(128x64)
  for (int tt = wave; tt < 32; tt += 8) {
    int it = tt >> 2, dt = tt & 3;
    v8f acc = {};
#pragma unroll
    for (int kc = 0; kc < 4; ++kc)
      acc = wmma16(frag_a_rm(P, LDN, it * 16, kc * 32),
                   frag_b_ct(&Vt[0][0], LDN, kc * 32, dt * 16), acc);
    int n  = dt * 16 + (lane & 15);
    int mb = it * 16 + ((lane >> 4) << 3);
#pragma unroll
    for (int r = 0; r < 8; ++r)
      o1[qb + (size_t)(mb + r) * DK_C + n] = acc[r];
  }
}

// ---------------------------------------------------------------------------
// K5: per (b,i): t[h][kk] = P_i(16x128) @ hdnV_i(128x64)   (hdnV stored [kk][j])
// ---------------------------------------------------------------------------
__global__ __launch_bounds__(128) void k5_t(
    const _Float16* __restrict__ hdnVt, const float* __restrict__ pattn,
    _Float16* __restrict__ tb) {
  const int i = blockIdx.x, b = blockIdx.y;
  const int t = threadIdx.x, lane = t & 31, wave = t >> 5;
  __shared__ __align__(16) _Float16 HVt[DK_C][LDN];  // [kk][j] (B^T: n=kk, k=j)
  __shared__ __align__(16) _Float16 Pt[H_C][LDN];    // [h][j]  (A)
  const size_t hb = (size_t)(b * N_C + i) * N_C * DK_C;

  // 16B packed copy: each row of HVt = 16 uint4
  for (int idx = t; idx < DK_C * N_C / 8; idx += 128) {
    int kk = idx >> 4, jp = idx & 15;
    ((uint4*)&HVt[kk][0])[jp] = ((const uint4*)(hdnVt + hb))[idx];
  }
  for (int idx = t; idx < H_C * N_C / 4; idx += 128) {
    int h = idx >> 5, j0 = (idx & 31) * 4;
    float4 p = ((const float4*)(pattn +
                ((size_t)(b * H_C + h) * N_C + i) * N_C))[idx & 31];
    Pt[h][j0 + 0] = (_Float16)p.x; Pt[h][j0 + 1] = (_Float16)p.y;
    Pt[h][j0 + 2] = (_Float16)p.z; Pt[h][j0 + 3] = (_Float16)p.w;
  }
  __syncthreads();

  {
    int kt = wave;  // 4 kk-tiles over 4 waves
    v8f acc = {};
#pragma unroll
    for (int kc = 0; kc < 4; ++kc)
      acc = wmma16(frag_a_rm(&Pt[0][0], LDN, 0, kc * 32),
                   frag_b_ct(&HVt[0][0], LDN, kc * 32, kt * 16), acc);
    int n  = kt * 16 + (lane & 15);       // kk
    int mb = (lane >> 4) << 3;            // h base
#pragma unroll
    for (int r = 0; r < 8; ++r)
      tb[((size_t)(b * H_C + (mb + r)) * N_C + i) * DK_C + n] = (_Float16)acc[r];
  }
}

// ---------------------------------------------------------------------------
// K6: per (b,h): out = t(128x64) @ Wv2_h^T(64x64) + out1 + bv2_h
// ---------------------------------------------------------------------------
__global__ __launch_bounds__(128) void k6_out(
    const _Float16* __restrict__ tb, const float* __restrict__ Wv2,
    const float* __restrict__ bv2, const float* __restrict__ o1,
    float* __restrict__ out) {
  const int h = blockIdx.x, b = blockIdx.y;
  const int t = threadIdx.x, lane = t & 31, wave = t >> 5;
  __shared__ __align__(16) _Float16 T[N_C][LDK];    // [i][kk]  (A)
  __shared__ __align__(16) _Float16 WD[DK_C][LDK];  // [d][kk]  (B^T: n=d, k=kk)
  const size_t ob = (size_t)(b * H_C + h) * N_C * DK_C;

  for (int idx = t; idx < N_C * DK_C / 8; idx += 128) {
    int i = idx >> 3, kp = idx & 7;
    ((uint4*)&T[i][0])[kp] = ((const uint4*)(tb + ob))[idx];
  }
  {
    const float4* w4 = (const float4*)(Wv2 + (size_t)h * DK_C * DK_C);
    for (int idx = t; idx < DK_C * DK_C / 4; idx += 128) {
      int d = idx >> 4, kk0 = (idx & 15) * 4;   // natural [d][kk]
      float4 w = w4[idx];
      WD[d][kk0 + 0] = (_Float16)w.x; WD[d][kk0 + 1] = (_Float16)w.y;
      WD[d][kk0 + 2] = (_Float16)w.z; WD[d][kk0 + 3] = (_Float16)w.w;
    }
  }
  __syncthreads();

  for (int tt = wave; tt < 32; tt += 4) {
    int it = tt >> 2, dt = tt & 3;
    v8f acc = {};
#pragma unroll
    for (int kc = 0; kc < 2; ++kc)
      acc = wmma16(frag_a_rm(&T[0][0], LDK, it * 16, kc * 32),
                   frag_b_ct(&WD[0][0], LDK, kc * 32, dt * 16), acc);
    int n  = dt * 16 + (lane & 15);
    int mb = it * 16 + ((lane >> 4) << 3);
#pragma unroll
    for (int r = 0; r < 8; ++r) {
      size_t idx = ob + (size_t)(mb + r) * DK_C + n;
      out[idx] = acc[r] + o1[idx] + bv2[h * DK_C + n];
    }
  }
}

// ---------------------------------------------------------------------------
extern "C" void kernel_launch(void* const* d_in, const int* in_sizes, int n_in,
                              void* d_out, int out_size, void* d_ws, size_t ws_size,
                              hipStream_t stream) {
  (void)in_sizes; (void)n_in; (void)out_size; (void)ws_size;
  const float* q     = (const float*)d_in[0];
  const float* key   = (const float*)d_in[1];
  const float* val   = (const float*)d_in[2];
  const float* mask  = (const float*)d_in[3];
  const float* edges = (const float*)d_in[4];
  const float* Wk1   = (const float*)d_in[5];
  const float* bk1   = (const float*)d_in[6];
  const float* Wk2   = (const float*)d_in[7];
  const float* bk2   = (const float*)d_in[8];
  const float* Wv1   = (const float*)d_in[9];
  const float* bv1   = (const float*)d_in[10];
  const float* Wv2   = (const float*)d_in[11];
  const float* bv2   = (const float*)d_in[12];
  const float* rel_u = (const float*)d_in[13];
  const float* rel_v = (const float*)d_in[14];

  float* out   = (float*)d_out;
  float* pattn = out + (size_t)B_C * H_C * N_C * DK_C;  // second output, flat

  // Workspace layout (~25 MB)
  char* ws = (char*)d_ws;
  _Float16* hdnK  = (_Float16*)(ws + (size_t)0);                  // 8 MB [b,i][j][k]
  _Float16* hdnVt = (_Float16*)(ws + (size_t)8  * 1024 * 1024);   // 8 MB [b,i][k][j]
  _Float16* gbuf  = (_Float16*)(ws + (size_t)16 * 1024 * 1024);   // 1 MB
  float*    cb    = (float*)   (ws + (size_t)17 * 1024 * 1024);   // 32 KB
  float*    s24   = (float*)   (ws + (size_t)18 * 1024 * 1024);   // 4 MB
  float*    o1    = (float*)   (ws + (size_t)22 * 1024 * 1024);   // 2 MB
  _Float16* tb    = (_Float16*)(ws + (size_t)24 * 1024 * 1024);   // 1 MB

  dim3 gbi(N_C, B_C);   // per (b, i)
  dim3 gbh(H_C, B_C);   // per (b, h)

  k1_edge_mlp<<<gbi, 128, 0, stream>>>(edges, Wk1, bk1, Wv1, bv1, hdnK, hdnVt);
  k2_g       <<<gbh, 128, 0, stream>>>(q, key, Wk2, bk2, rel_u, rel_v, gbuf, cb);
  k3_s24     <<<gbi, 128, 0, stream>>>(hdnK, gbuf, cb, s24);
  k4_attn    <<<gbh, 256, 0, stream>>>(q, key, val, mask, s24, o1, pattn);
  k5_t       <<<gbi, 128, 0, stream>>>(hdnVt, pattn, tb);
  k6_out     <<<gbh, 128, 0, stream>>>(tb, Wv2, bv2, o1, out);
}
